// LIF_8933531976421
// MI455X (gfx1250) — compile-verified, matrix operand order checked by reading, not run on hardware
//
#include <hip/hip_runtime.h>

// LIF recurrent spiking net forward for MI455X (gfx1250, wave32, WMMA).
//  k0: f32->f16 convert (W, R)
//  k1: init z0=0, barrier=0
//  k2: input GEMM current = x @ W^T + bias (f16 WMMA, f32 accum,
//      4 n-tiles per wave to cut x re-reads 4.2GB -> ~1GB)
//  k3: persistent LIF scan, 1000 steps, register-resident R fragments,
//      register-resident membrane potential, 4 independent WMMA accumulation
//      chains (latency), atomic grid barrier per step.

typedef __attribute__((ext_vector_type(16))) _Float16 v16h;
typedef __attribute__((ext_vector_type(8)))  _Float16 v8h;
typedef __attribute__((ext_vector_type(8)))  float    v8f;
typedef __attribute__((ext_vector_type(4)))  float    v4f;

#define B_  64
#define T_  1000
#define N_  512
#define IN_ 512

// ---------------- workspace layout (bytes) ----------------
// [0, CURB)     : current f32  [B,T,N]   131,072,000
// [+WHB)        : W  f16 [N,IN]              524,288
// [+RHB)        : R  f16 [N,N]               524,288
// [+ZB)         : z ping [B,N] f16            65,536
// [+ZB)         : z pong [B,N] f16            65,536
// [+4)          : grid-barrier counter
#define CURB ((size_t)B_ * T_ * N_ * 4)
#define WHB  ((size_t)N_ * IN_ * 2)
#define RHB  ((size_t)N_ * N_ * 2)
#define ZB   ((size_t)B_ * N_ * 2)

// ---------------- f32 -> f16 convert ----------------
__global__ void __launch_bounds__(256) cvt_f32_f16(const float* __restrict__ s,
                                                   _Float16* __restrict__ d, int n) {
  int i = blockIdx.x * blockDim.x + threadIdx.x;
  int stride = gridDim.x * blockDim.x;
  for (; i < n; i += stride) d[i] = (_Float16)s[i];
}

// ---------------- init z0 = 0, barrier = 0 ----------------
__global__ void __launch_bounds__(256) scan_init(_Float16* __restrict__ z0,
                                                 unsigned* __restrict__ bar, int n) {
  int i = blockIdx.x * blockDim.x + threadIdx.x;
  if (i < n) z0[i] = (_Float16)0.0f;
  if (i == 0) *bar = 0u;
}

// ---------------- input projection GEMM ----------------
// current[m][n] = sum_k x[m][k] * W[n][k] + bias[n]
// One wave computes a 16(M) x 64(N) strip: 4 n-tiles sharing one A fragment.
__global__ void __launch_bounds__(256) input_gemm(const float* __restrict__ x,
                                                  const _Float16* __restrict__ Wh,
                                                  const float* __restrict__ bias,
                                                  float* __restrict__ cur) {
  const int lane = threadIdx.x & 31;
  const int warp = threadIdx.x >> 5;
  const int wid  = blockIdx.x * 8 + warp;   // 0 .. 31999
  const int m_tile = wid >> 3;              // 0 .. 3999
  const int ngrp   = wid & 7;               // 0 .. 7  (64-col groups)
  const int l15  = lane & 15;
  const int kb   = (lane >> 4) << 3;        // 0 or 8 (A/B K-half select)
  const int moff = (lane >> 4) << 3;        // C/D M-half select

  const float* xrow = x + (size_t)((m_tile << 4) + l15) * IN_;

  int ncol[4];
  const _Float16* wrow[4];
#pragma unroll
  for (int j = 0; j < 4; ++j) {
    ncol[j] = (ngrp << 6) + (j << 4) + l15;
    wrow[j] = Wh + (size_t)ncol[j] * IN_;
  }

  v8f acc[4] = {};
  for (int k0 = 0; k0 < IN_; k0 += 32) {
    // A fragment: x row, f32 -> f16 on the fly (4x b128 loads + pk cvt)
    v4f x0 = *(const v4f*)(xrow + k0 + kb);
    v4f x1 = *(const v4f*)(xrow + k0 + kb + 4);
    v4f x2 = *(const v4f*)(xrow + k0 + kb + 16);
    v4f x3 = *(const v4f*)(xrow + k0 + kb + 20);
    v16h a;
#pragma unroll
    for (int i = 0; i < 4; ++i) {
      a[i]      = (_Float16)x0[i];
      a[4 + i]  = (_Float16)x1[i];
      a[8 + i]  = (_Float16)x2[i];
      a[12 + i] = (_Float16)x3[i];
    }
    // 4 B fragments (W rows, contiguous f16) -> 4 independent WMMAs
#pragma unroll
    for (int j = 0; j < 4; ++j) {
      v8h b0 = *(const v8h*)(wrow[j] + k0 + kb);
      v8h b1 = *(const v8h*)(wrow[j] + k0 + kb + 16);
      v16h b;
#pragma unroll
      for (int i = 0; i < 8; ++i) { b[i] = b0[i]; b[8 + i] = b1[i]; }
      acc[j] = __builtin_amdgcn_wmma_f32_16x16x32_f16(false, a, false, b,
                                                      (short)0, acc[j], false, false);
    }
  }

#pragma unroll
  for (int j = 0; j < 4; ++j) {
    const float bs = bias[ncol[j]];
#pragma unroll
    for (int r = 0; r < 8; ++r) {
      int m = (m_tile << 4) + moff + r;
      cur[(size_t)m * N_ + ncol[j]] = acc[j][r] + bs;
    }
  }
}

// ---------------- persistent LIF scan ----------------
// 16 blocks x 8 waves = 128 waves = 4 (batch) x 32 (neuron) 16x16 tiles.
__global__ void __launch_bounds__(256) lif_scan(const float* __restrict__ cur,
                                                const _Float16* __restrict__ Rh,
                                                const float* __restrict__ decay,
                                                const float* __restrict__ thr,
                                                const float* __restrict__ u0,
                                                _Float16* __restrict__ zA,
                                                _Float16* __restrict__ zB,
                                                float* __restrict__ out,
                                                unsigned* __restrict__ bar) {
  const int lane = threadIdx.x & 31;
  const int warp = threadIdx.x >> 5;
  const int wid  = blockIdx.x * 8 + warp;   // 0 .. 127
  const int m_tile = wid >> 5;              // 0 .. 3   (batch tiles)
  const int n_tile = wid & 31;              // 0 .. 31  (neuron tiles)
  const int l15  = lane & 15;
  const int kb   = (lane >> 4) << 3;
  const int moff = (lane >> 4) << 3;
  const int ncol = (n_tile << 4) + l15;
  const int arow = (m_tile << 4) + l15;     // batch row this lane loads for A

  // Register-resident B fragments of R for this neuron tile: live all 1000 steps.
  const _Float16* rrow = Rh + (size_t)ncol * N_;
  v16h Bf[16];
#pragma unroll
  for (int kk = 0; kk < 16; ++kk) {
    const int base = kk * 32 + kb;
    v8h r0 = *(const v8h*)(rrow + base);
    v8h r1 = *(const v8h*)(rrow + base + 16);
#pragma unroll
    for (int i = 0; i < 8; ++i) { Bf[kk][i] = r0[i]; Bf[kk][8 + i] = r1[i]; }
  }

  // Register-resident membrane potential for the 8 (b,n) cells this lane owns.
  const float dec = decay[ncol];
  const float th  = thr[0];
  float u[8];
  {
    const float ui = u0[ncol];
#pragma unroll
    for (int r = 0; r < 8; ++r) u[r] = ui;
  }

  const unsigned nblk = gridDim.x;
  for (int t = 0; t < T_; ++t) {
    const _Float16* zprev = (t & 1) ? zB : zA;
    _Float16*       znext = (t & 1) ? zA : zB;

    // Issue current[t] loads first so their latency overlaps the WMMA chain.
    float c[8];
#pragma unroll
    for (int r = 0; r < 8; ++r) {
      const int b = (m_tile << 4) + moff + r;
      c[r] = cur[((size_t)b * T_ + t) * N_ + ncol];
    }

    // rec = z_{t-1} @ R^T tile: 4 independent WMMA chains of depth 4
    // (breaks the serial C-operand dependency; scan is latency-bound).
    v8f a0 = {}, a1 = {}, a2 = {}, a3 = {};
    const _Float16* zrow = zprev + (size_t)arow * N_;
#pragma unroll
    for (int kk = 0; kk < 4; ++kk) {
#pragma unroll
      for (int cchain = 0; cchain < 4; ++cchain) {
        const int kidx = kk * 4 + cchain;
        const int base = kidx * 32 + kb;
        v8h z0 = *(const v8h*)(zrow + base);
        v8h z1 = *(const v8h*)(zrow + base + 16);
        v16h a;
#pragma unroll
        for (int i = 0; i < 8; ++i) { a[i] = z0[i]; a[8 + i] = z1[i]; }
        v8f* accp = (cchain == 0) ? &a0 : (cchain == 1) ? &a1 : (cchain == 2) ? &a2 : &a3;
        *accp = __builtin_amdgcn_wmma_f32_16x16x32_f16(false, a, false, Bf[kidx],
                                                       (short)0, *accp, false, false);
      }
    }

    // LIF update + spike, write z to ping-pong (f16) and output (f32)
#pragma unroll
    for (int r = 0; r < 8; ++r) {
      const int b = (m_tile << 4) + moff + r;
      const size_t co = ((size_t)b * T_ + t) * N_ + ncol;
      const float rec = (a0[r] + a1[r]) + (a2[r] + a3[r]);
      const float cc  = c[r] + rec;
      const float un  = dec * u[r] + (1.0f - dec) * cc;
      const float z   = (un - th > 0.0f) ? 1.0f : 0.0f;
      u[r] = un * (1.0f - z);
      znext[(size_t)b * N_ + ncol] = (_Float16)z;
      out[co] = z;
    }

    // Prefetch next step's current slice (global_prefetch_b8)
    if (t + 1 < T_) {
#pragma unroll
      for (int r = 0; r < 8; ++r) {
        const int b = (m_tile << 4) + moff + r;
        __builtin_prefetch(cur + ((size_t)b * T_ + (t + 1)) * N_ + ncol, 0, 0);
      }
    }

    // Grid barrier: monotonic counter, 16 co-resident blocks.
    __syncthreads();
    if (threadIdx.x == 0) {
      __threadfence();
      atomicAdd(bar, 1u);
      const unsigned target = (unsigned)(t + 1) * nblk;
      while (__hip_atomic_load(bar, __ATOMIC_ACQUIRE, __HIP_MEMORY_SCOPE_AGENT) < target) {
      }
    }
    __syncthreads();
  }
}

extern "C" void kernel_launch(void* const* d_in, const int* in_sizes, int n_in,
                              void* d_out, int out_size, void* d_ws, size_t ws_size,
                              hipStream_t stream) {
  const float* x     = (const float*)d_in[0];
  const float* W     = (const float*)d_in[1];
  const float* bias  = (const float*)d_in[2];
  const float* R     = (const float*)d_in[3];
  const float* decay = (const float*)d_in[4];
  const float* thr   = (const float*)d_in[5];
  const float* u0    = (const float*)d_in[6];
  float* out = (float*)d_out;

  char* ws = (char*)d_ws;                    // needs ~132.3 MB
  float*     cur = (float*)ws;
  _Float16*  Wh  = (_Float16*)(ws + CURB);
  _Float16*  Rh  = (_Float16*)(ws + CURB + WHB);
  _Float16*  z0  = (_Float16*)(ws + CURB + WHB + RHB);
  _Float16*  z1  = (_Float16*)(ws + CURB + WHB + RHB + ZB);
  unsigned*  bar = (unsigned*)(ws + CURB + WHB + RHB + ZB + ZB);

  cvt_f32_f16<<<512, 256, 0, stream>>>(W, Wh, N_ * IN_);
  cvt_f32_f16<<<512, 256, 0, stream>>>(R, Rh, N_ * N_);
  scan_init<<<(B_ * N_ + 255) / 256, 256, 0, stream>>>(z0, bar, B_ * N_);

  // 4000 M tiles x 8 n-groups (64 cols each) = 32000 waves / 8 per block
  input_gemm<<<4000, 256, 0, stream>>>(x, Wh, bias, cur);

  // persistent scan: 16 blocks x 256 threads (co-resident; grid barrier safe)
  lif_scan<<<16, 256, 0, stream>>>(cur, Rh, decay, thr, u0, z0, z1, out, bar);
}
